// Beit_68882685494028
// MI455X (gfx1250) — matrix operand order repaired and checked by reading, not compiled, not run on hardware
//
#include <hip/hip_runtime.h>
#include <hip/hip_bf16.h>
#include <stdint.h>

#define DEV __device__ __forceinline__

typedef __bf16 bf16_t;
typedef __attribute__((ext_vector_type(16))) __bf16 v16bf;
typedef __attribute__((ext_vector_type(8)))  float  v8f;
typedef __attribute__((ext_vector_type(4)))  unsigned int u32x4;
typedef __attribute__((ext_vector_type(8)))  int i32x8;
typedef __attribute__((ext_vector_type(4)))  int i32x4;

constexpr int   kDim   = 768;
constexpr int   kHeads = 12;
constexpr int   kTok   = 197;
constexpr int   kHd    = 64;
constexpr int   kBatch = 256;
constexpr int   kM     = kBatch * kTok;   // 50432 = 256 * 197
constexpr float kScale = 0.125f;          // 64^-0.5
constexpr int   kNPad  = 208;             // 13 * 16
constexpr int   kKPad  = 224;             // 7 * 32
constexpr int   kKT    = kDim / 32;       // 24 K-tiles per GEMM
constexpr int   kBM    = 256;             // GEMM M-tile (8 waves x 32 rows)

DEV uint16_t f2bf(float f) {
  union { float f; uint32_t u; } c; c.f = f;
  return (uint16_t)((c.u + 0x7FFFu + ((c.u >> 16) & 1u)) >> 16);
}

union Frag16 { uint4 q[2]; v16bf v; };

// A-matrix fragment (16x32 bf16): lanes 0-15 hold K 0..7 / 16..23,
// lanes 16-31 hold K 8..15 / 24..31 (ISA 7.12.2).
DEV v16bf load_a(const bf16_t* rowptr, int lane) {
  const int kh = (lane < 16) ? 0 : 8;
  Frag16 f;
  f.q[0] = *reinterpret_cast<const uint4*>(rowptr + kh);
  f.q[1] = *reinterpret_cast<const uint4*>(rowptr + kh + 16);
  return f.v;
}

// B-matrix fragment (32x16 bf16): lanes 0-15 hold K 0..7 / 16..23,
// lanes 16-31 hold K 8..15 / 24..31 (same order as A).
DEV v16bf load_b(const bf16_t* rowptr, int lane) {
  const int kh = (lane < 16) ? 0 : 8;
  Frag16 f;
  f.q[0] = *reinterpret_cast<const uint4*>(rowptr + kh);
  f.q[1] = *reinterpret_cast<const uint4*>(rowptr + kh + 16);
  return f.v;
}

DEV v8f wmma_bf16(v16bf a, v16bf b, v8f c) {
  return __builtin_amdgcn_wmma_f32_16x16x32_bf16(false, a, false, b, (short)0, c,
                                                 false, false);
}

// ---- Tensor Data Mover: 2D bf16 tile, global -> LDS (ISA ch.8 D# layout) ----
DEV void tdm_load_2d(void* lds_dst, const void* gsrc,
                     uint32_t width, uint32_t src_rows, uint32_t dst_rows,
                     uint32_t stride_elems) {
  const uint64_t ga = (uint64_t)(uintptr_t)gsrc;
  const uint32_t la = (uint32_t)(uintptr_t)lds_dst;   // flat addr[31:0] = LDS addr
  u32x4 g0;
  g0[0] = 1u;                                          // count=1, is_restore=0
  g0[1] = la;                                          // lds_addr
  g0[2] = (uint32_t)ga;                                // global_addr[31:0]
  g0[3] = ((uint32_t)(ga >> 32) & 0x01FFFFFFu) | 0x80000000u; // ga[56:32], type=2
  uint32_t d[8];
  d[0] = 1u << 16;                                     // data_size=1 (2 bytes)
  d[1] = (width & 0xFFFFu) << 16;                      // tensor_dim0[15:0]
  d[2] = ((width >> 16) & 0xFFFFu) | ((src_rows & 0xFFFFu) << 16);
  d[3] = ((src_rows >> 16) & 0xFFFFu) | ((width & 0xFFFFu) << 16); // tile_dim0
  d[4] = dst_rows & 0xFFFFu;                           // tile_dim1 (tile_dim2=0)
  d[5] = stride_elems;                                 // tensor_dim0_stride[31:0]
  d[6] = 0u;
  d[7] = 0u;
  i32x8 g1 = {(int)d[0], (int)d[1], (int)d[2], (int)d[3],
              (int)d[4], (int)d[5], (int)d[6], (int)d[7]};
  i32x4 z4 = {0, 0, 0, 0};
#if __clang_major__ >= 23
  i32x8 z8 = {0, 0, 0, 0, 0, 0, 0, 0};
  __builtin_amdgcn_tensor_load_to_lds(g0, g1, z4, z4, z8, 0);
#else
  __builtin_amdgcn_tensor_load_to_lds(g0, g1, z4, z4, 0);
#endif
}

// ---------------- fp32 -> bf16 bulk convert (8 elems / thread) ----------------
__global__ void cvt8_f32_bf16(const float* __restrict__ src,
                              bf16_t* __restrict__ dst, long long n8) {
  long long i = (long long)blockIdx.x * blockDim.x + threadIdx.x;
  if (i >= n8) return;
  const float4* s = reinterpret_cast<const float4*>(src) + i * 2;
  float4 a = s[0], b = s[1];
  uint4 o;
  o.x = (uint32_t)f2bf(a.x) | ((uint32_t)f2bf(a.y) << 16);
  o.y = (uint32_t)f2bf(a.z) | ((uint32_t)f2bf(a.w) << 16);
  o.z = (uint32_t)f2bf(b.x) | ((uint32_t)f2bf(b.y) << 16);
  o.w = (uint32_t)f2bf(b.z) | ((uint32_t)f2bf(b.w) << 16);
  reinterpret_cast<uint4*>(dst)[i] = o;
}

// ---------------- gather rel-pos bias into [H, 197, 197] f32 ----------------
__global__ void gather_bias(const float* __restrict__ table,
                            const int* __restrict__ ridx,
                            float* __restrict__ biasG) {
  int i = blockIdx.x * 256 + threadIdx.x;
  if (i >= kHeads * kTok * kTok) return;
  int h = i / (kTok * kTok);
  int r = i - h * (kTok * kTok);
  biasG[i] = table[ridx[r] * kHeads + h];
}

// ---- shared GEMM inner machinery: 256x64 block, wave = 32x64, TDM staged ----
// acc[2][4]; Xs tile [256][32], Ws tile [64][32], double buffered.
#define GEMM_BODY(APTR, WPTR)                                                   \
  __shared__ bf16_t Xs[2][kBM][32];   /* 32 KB */                               \
  __shared__ bf16_t Ws[2][64][32];    /*  8 KB */                               \
  const int tid = threadIdx.x, lane = tid & 31, wv = tid >> 5;                  \
  const int n0 = blockIdx.x * 64;                                               \
  const int m0 = blockIdx.y * kBM;                                              \
  v8f acc[2][4];                                                                \
  _Pragma("unroll") for (int rb = 0; rb < 2; ++rb)                              \
  _Pragma("unroll") for (int nt = 0; nt < 4; ++nt) { v8f z = {}; acc[rb][nt] = z; } \
  if (wv == 0) {                                                                \
    tdm_load_2d(&Xs[0][0][0], &(APTR)[(size_t)m0 * kDim], 32, kBM, kBM, kDim);  \
    tdm_load_2d(&Ws[0][0][0], &(WPTR)[(size_t)n0 * kDim], 32, 64, 64, kDim);    \
  }                                                                             \
  for (int t = 0; t < kKT; ++t) {                                               \
    const int cur = t & 1;                                                      \
    if (wv == 0) {                                                              \
      if (t + 1 < kKT) {                                                        \
        const int k1 = (t + 1) * 32;                                            \
        tdm_load_2d(&Xs[cur ^ 1][0][0], &(APTR)[(size_t)m0 * kDim + k1], 32, kBM, kBM, kDim); \
        tdm_load_2d(&Ws[cur ^ 1][0][0], &(WPTR)[(size_t)n0 * kDim + k1], 32, 64, 64, kDim);   \
        __builtin_amdgcn_s_wait_tensorcnt(2);                                   \
      } else {                                                                  \
        __builtin_amdgcn_s_wait_tensorcnt(0);                                   \
      }                                                                         \
    }                                                                           \
    __syncthreads();                                                            \
    v16bf a0 = load_a(&Xs[cur][wv * 32 + (lane & 15)][0], lane);                \
    v16bf a1 = load_a(&Xs[cur][wv * 32 + 16 + (lane & 15)][0], lane);           \
    v16bf b0 = load_b(&Ws[cur][0 + (lane & 15)][0], lane);                      \
    v16bf b1 = load_b(&Ws[cur][16 + (lane & 15)][0], lane);                     \
    v16bf b2 = load_b(&Ws[cur][32 + (lane & 15)][0], lane);                     \
    v16bf b3 = load_b(&Ws[cur][48 + (lane & 15)][0], lane);                     \
    acc[0][0] = wmma_bf16(a0, b0, acc[0][0]);                                   \
    acc[0][1] = wmma_bf16(a0, b1, acc[0][1]);                                   \
    acc[0][2] = wmma_bf16(a0, b2, acc[0][2]);                                   \
    acc[0][3] = wmma_bf16(a0, b3, acc[0][3]);                                   \
    acc[1][0] = wmma_bf16(a1, b0, acc[1][0]);                                   \
    acc[1][1] = wmma_bf16(a1, b1, acc[1][1]);                                   \
    acc[1][2] = wmma_bf16(a1, b2, acc[1][2]);                                   \
    acc[1][3] = wmma_bf16(a1, b3, acc[1][3]);                                   \
    __syncthreads();                                                            \
  }

// ---------------- QKV GEMM, fused bias/scale, bf16 out per head ------------
__global__ __launch_bounds__(256) void qkv_gemm(
    const bf16_t* __restrict__ Xb, const bf16_t* __restrict__ Wb,
    const float* __restrict__ qbias, const float* __restrict__ vbias,
    bf16_t* __restrict__ Q, bf16_t* __restrict__ K, bf16_t* __restrict__ V) {
  GEMM_BODY(Xb, Wb)

  const int region = n0 / kDim;            // 0=q, 1=k, 2=v
  const int cin0 = n0 - region * kDim;
  const int head = cin0 / kHd;             // block spans exactly one head
  bf16_t* dst = (region == 0) ? Q : (region == 1) ? K : V;
  const float scl = (region == 0) ? kScale : 1.0f;
  const int half = (lane >= 16) ? 8 : 0;
#pragma unroll
  for (int rb = 0; rb < 2; ++rb) {
#pragma unroll
    for (int nt = 0; nt < 4; ++nt) {
      const int d = nt * 16 + (lane & 15);   // 0..63
      float bias = 0.0f;
      if (region == 0) bias = qbias[cin0 + d];
      else if (region == 2) bias = vbias[cin0 + d];
#pragma unroll
      for (int vg = 0; vg < 8; ++vg) {
        int m = m0 + wv * 32 + rb * 16 + vg + half;
        int bb = m / kTok, tok = m - bb * kTok;
        size_t o = (((size_t)bb * kHeads + head) * kTok + tok) * kHd + d;
        reinterpret_cast<uint16_t*>(dst)[o] = f2bf((acc[rb][nt][vg] + bias) * scl);
      }
    }
  }
}

// ---------------- attention: one block per (batch, head) -------------------
__global__ __launch_bounds__(256) void attn_kernel(
    const bf16_t* __restrict__ Q, const bf16_t* __restrict__ K,
    const bf16_t* __restrict__ V, const float* __restrict__ biasG,
    bf16_t* __restrict__ Ob) {
  __shared__ bf16_t Ks[kNPad][64];    // k row-major   (26.0 KB)
  __shared__ bf16_t Vt[64][kKPad];    // v transposed  (28.0 KB)
  __shared__ bf16_t Ps[8][16][32];    // per-wave P restage (8 KB)
  const int tid = threadIdx.x, lane = tid & 31, wv = tid >> 5;
  const int bh = blockIdx.x;
  const int bb = bh / kHeads, h = bh - bb * kHeads;
  const size_t base = (size_t)bh * kTok * kHd;

  // K tile via TDM (197 valid rows; rows 197..207 zero-filled by OOB reads),
  // overlapped with the manual V transpose scatter below.
  if (wv == 0)
    tdm_load_2d(&Ks[0][0], &K[base], kHd, kTok, kNPad, kHd);

  for (int i = tid; i < kNPad * 8; i += 256) {
    int tok = i >> 3, cc = (i & 7) << 3;
    uint32_t w[4] = {0, 0, 0, 0};
    if (tok < kTok) {
      uint4 u = *reinterpret_cast<const uint4*>(&V[base + (size_t)tok * kHd + cc]);
      w[0] = u.x; w[1] = u.y; w[2] = u.z; w[3] = u.w;
    }
#pragma unroll
    for (int j = 0; j < 4; ++j) {
      reinterpret_cast<uint16_t*>(&Vt[cc + 2 * j][tok])[0] = (uint16_t)(w[j] & 0xFFFFu);
      reinterpret_cast<uint16_t*>(&Vt[cc + 2 * j + 1][tok])[0] = (uint16_t)(w[j] >> 16);
    }
  }
  for (int i = tid; i < 64 * 2; i += 256) {   // zero Vt columns 208..223
    int row = i >> 1, cc = 208 + ((i & 1) << 3);
    uint4 z = {0, 0, 0, 0};
    *reinterpret_cast<uint4*>(&Vt[row][cc]) = z;
  }
  if (wv == 0) __builtin_amdgcn_s_wait_tensorcnt(0);
  __syncthreads();

  const int half = (lane >= 16) ? 8 : 0;
  const int col = lane & 15;

  for (int s = wv; s < 13; s += 8) {          // 16-row q slabs
    v16bf qa[2];
    {
      int row = s * 16 + (lane & 15);
      if (row > kTok - 1) row = kTok - 1;     // clamp pad rows (discarded later)
      const bf16_t* qp = &Q[base + (size_t)row * kHd];
      qa[0] = load_a(qp, lane);
      qa[1] = load_a(qp + 32, lane);
    }
    v8f S[13];
#pragma unroll
    for (int kt = 0; kt < 13; ++kt) { v8f z = {}; S[kt] = z; }
#pragma unroll
    for (int kk = 0; kk < 2; ++kk) {
#pragma unroll
      for (int kt = 0; kt < 13; ++kt) {
        v16bf b = load_b(&Ks[kt * 16 + (lane & 15)][kk * 32], lane);
        S[kt] = wmma_bf16(qa[kk], b, S[kt]);
      }
    }
    // rel-pos bias + pad-column masking
#pragma unroll
    for (int kt = 0; kt < 13; ++kt) {
      int kc = kt * 16 + col;
#pragma unroll
      for (int vg = 0; vg < 8; ++vg) {
        int qr = s * 16 + vg + half;
        if (qr > kTok - 1) qr = kTok - 1;
        float bv = (kc < kTok) ? biasG[((size_t)h * kTok + qr) * kTok + kc]
                               : -1.0e30f;
        S[kt][vg] += bv;
      }
    }
    // row softmax (16-lane reductions; halves hold distinct rows)
    float inv[8];
#pragma unroll
    for (int vg = 0; vg < 8; ++vg) {
      float mx = -3.0e38f;
#pragma unroll
      for (int kt = 0; kt < 13; ++kt) mx = fmaxf(mx, S[kt][vg]);
#pragma unroll
      for (int off = 1; off < 16; off <<= 1) mx = fmaxf(mx, __shfl_xor(mx, off, 32));
      float sum = 0.0f;
#pragma unroll
      for (int kt = 0; kt < 13; ++kt) {
        float e = __expf(S[kt][vg] - mx);
        S[kt][vg] = e;
        sum += e;
      }
#pragma unroll
      for (int off = 1; off < 16; off <<= 1) sum += __shfl_xor(sum, off, 32);
      inv[vg] = 1.0f / sum;
    }
    // O = P @ V, K restaged 32 tokens at a time through per-wave LDS tile
    v8f O[4];
#pragma unroll
    for (int dt = 0; dt < 4; ++dt) { v8f z = {}; O[dt] = z; }
    for (int c = 0; c < 7; ++c) {
#pragma unroll
      for (int t = 0; t < 2; ++t) {
        int kt = c * 2 + t;
#pragma unroll
        for (int vg = 0; vg < 8; ++vg) {
          float p = (kt < 13) ? S[kt][vg] * inv[vg] : 0.0f;
          reinterpret_cast<uint16_t*>(&Ps[wv][vg + half][t * 16 + col])[0] = f2bf(p);
        }
      }
      asm volatile("s_wait_dscnt 0" ::: "memory");  // wave-local LDS fence
      v16bf a = load_a(&Ps[wv][lane & 15][0], lane);
#pragma unroll
      for (int dt = 0; dt < 4; ++dt) {
        v16bf b = load_b(&Vt[dt * 16 + (lane & 15)][c * 32], lane);
        O[dt] = wmma_bf16(a, b, O[dt]);
      }
    }
    // store O as bf16 [B, N, DIM] (proj GEMM input layout)
#pragma unroll
    for (int dt = 0; dt < 4; ++dt) {
#pragma unroll
      for (int vg = 0; vg < 8; ++vg) {
        int qr = s * 16 + vg + half;
        if (qr < kTok) {
          size_t o = ((size_t)bb * kTok + qr) * kDim + h * kHd + dt * 16 + col;
          reinterpret_cast<uint16_t*>(Ob)[o] = f2bf(O[dt][vg]);
        }
      }
    }
  }
}

// ---------------- proj GEMM, fp32 out + bias ------------------------------
__global__ __launch_bounds__(256) void proj_gemm(
    const bf16_t* __restrict__ Ob, const bf16_t* __restrict__ Wp,
    const float* __restrict__ pbias, float* __restrict__ out) {
  GEMM_BODY(Ob, Wp)

  const int half = (lane >= 16) ? 8 : 0;
#pragma unroll
  for (int rb = 0; rb < 2; ++rb) {
#pragma unroll
    for (int nt = 0; nt < 4; ++nt) {
      const int c = n0 + nt * 16 + (lane & 15);
      const float bias = pbias[c];
#pragma unroll
      for (int vg = 0; vg < 8; ++vg) {
        int m = m0 + wv * 32 + rb * 16 + vg + half;
        out[(size_t)m * kDim + c] = acc[rb][nt][vg] + bias;
      }
    }
  }
}

// ---------------- host-side launch --------------------------------------
static constexpr size_t alignUp(size_t v) { return (v + 255) & ~(size_t)255; }

extern "C" void kernel_launch(void* const* d_in, const int* in_sizes, int n_in,
                              void* d_out, int out_size, void* d_ws, size_t ws_size,
                              hipStream_t stream) {
  const float* x      = (const float*)d_in[0];
  const float* qkv_w  = (const float*)d_in[1];
  const float* q_bias = (const float*)d_in[2];
  const float* v_bias = (const float*)d_in[3];
  const float* table  = (const float*)d_in[4];
  const float* proj_w = (const float*)d_in[5];
  const float* proj_b = (const float*)d_in[6];
  const int*   ridx   = (const int*)d_in[7];
  float* out = (float*)d_out;

  constexpr size_t SZ_XB   = (size_t)kM * kDim * 2;                      // bf16 x
  constexpr size_t SZ_WQKV = (size_t)3 * kDim * kDim * 2;                // bf16 qkv_w
  constexpr size_t SZ_WPRJ = (size_t)kDim * kDim * 2;                    // bf16 proj_w
  constexpr size_t SZ_HEAD = (size_t)kBatch * kHeads * kTok * kHd * 2;   // bf16 q/k/v
  constexpr size_t SZ_BIAS = (size_t)kHeads * kTok * kTok * 4;           // f32 bias

  char* w = (char*)d_ws;
  size_t off = 0;
  bf16_t* Xb = (bf16_t*)(w + off); off += alignUp(SZ_XB);
  bf16_t* Wq = (bf16_t*)(w + off); off += alignUp(SZ_WQKV);
  bf16_t* Wp = (bf16_t*)(w + off); off += alignUp(SZ_WPRJ);
  bf16_t* Qb = (bf16_t*)(w + off); off += alignUp(SZ_HEAD);
  bf16_t* Kb = (bf16_t*)(w + off); off += alignUp(SZ_HEAD);
  bf16_t* Vb = (bf16_t*)(w + off); off += alignUp(SZ_HEAD);
  float*  Bg = (float*)(w + off);  off += alignUp(SZ_BIAS);
  bf16_t* Ob = (bf16_t*)(w + off); off += alignUp(SZ_XB);
  (void)ws_size; (void)in_sizes; (void)n_in; (void)out_size;

  {
    long long n8 = (long long)kM * kDim / 8;
    cvt8_f32_bf16<<<(unsigned)((n8 + 255) / 256), 256, 0, stream>>>(x, Xb, n8);
  }
  {
    long long n8 = (long long)3 * kDim * kDim / 8;
    cvt8_f32_bf16<<<(unsigned)((n8 + 255) / 256), 256, 0, stream>>>(qkv_w, Wq, n8);
  }
  {
    long long n8 = (long long)kDim * kDim / 8;
    cvt8_f32_bf16<<<(unsigned)((n8 + 255) / 256), 256, 0, stream>>>(proj_w, Wp, n8);
  }
  {
    int n = kHeads * kTok * kTok;
    gather_bias<<<(n + 255) / 256, 256, 0, stream>>>(table, ridx, Bg);
  }
  qkv_gemm<<<dim3(3 * kDim / 64, kM / kBM), 256, 0, stream>>>(
      Xb, Wq, q_bias, v_bias, Qb, Kb, Vb);
  attn_kernel<<<kBatch * kHeads, 256, 0, stream>>>(Qb, Kb, Vb, Bg, Ob);
  proj_gemm<<<dim3(kDim / 64, kM / kBM), 256, 0, stream>>>(Ob, Wp, proj_b, out);
}
